// CRNN_decoder_69037304316020
// MI455X (gfx1250) — matrix operand order repaired and checked by reading, not compiled, not run on hardware
//
#include <hip/hip_runtime.h>

typedef __bf16 bf16;
typedef __attribute__((ext_vector_type(16))) __bf16 v16bf;
typedef __attribute__((ext_vector_type(8)))  float  v8f;

// ---------------- static dims ----------------
constexpr int B_   = 16;
constexpr int T_   = 5760;        // Y*M*WIN
constexpr int DENC = 256;
constexpr int U_   = 128;
constexpr int A1_  = 64;
constexpr int A2_  = 32;
constexpr int RNN  = 256;
constexpr int FDEC = 64;
constexpr int NOUT = 64;
constexpr int FLATM = 40992;      // 42*122*8   (= 32*1281)
constexpr int FLATY = 524112;     // 537*122*8
constexpr int FLATY_PAD = 524128; // padded to multiple of 32 (32*16379)
constexpr int YSPLIT = 32;        // K-splits for yearly dense

union PK4 { bf16 h[4]; uint2 u; };

// ---------------- WMMA helpers ----------------
__device__ inline v8f wmma_bf16(v16bf a, v16bf b, v8f c) {
  return __builtin_amdgcn_wmma_f32_16x16x32_bf16(false, a, false, b, (short)0, c,
                                                 false, false);
}

// A fragment (16x32 bf16) from LDS, row stride lda.
// lanes 0-15: M=lane, elems 0..15 -> K 0..15
// lanes 16-31: M=lane-16, elems 0..15 -> K 16..31
__device__ inline v16bf load_a_frag(const bf16* s, int lda, int kbase, int lane) {
  int m  = lane & 15;
  int kh = (lane >> 4) << 4;  // 0 or 16
  const bf16* p = s + m * lda + kbase + kh;
  v16bf a;
#pragma unroll
  for (int e = 0; e < 16; ++e) a[e] = p[e];
  return a;
}

// B fragment from pre-packed weights: layout [K/32][N/16][32 lanes][16 bf16].
// One contiguous 32-byte load per lane (2x global_load_b128).
__device__ inline v16bf load_b_packed(const bf16* g, int ntiles, int tk, int tn,
                                      int lane) {
  const bf16* p = g + ((((size_t)tk * ntiles + tn) << 5) + lane) * 16;
  return *(const v16bf*)p;
}

// ---------------- pack f32 [K][N] weights into WMMA-B fragment order ----------------
// dst[tk][tn][lane][e] = W[tk*32 + (lane>=16)*16 + e][tn*16 + (lane&15)], zero-padded K
__global__ void k_pack_b(const float* __restrict__ src, bf16* __restrict__ dst,
                         int K, int N, int Kpad) {
  const int ntiles = N >> 4;
  const long long total = (long long)(Kpad >> 5) * ntiles * 512;
  long long idx = (long long)blockIdx.x * blockDim.x + threadIdx.x;
  const long long stride = (long long)gridDim.x * blockDim.x;
  for (; idx < total; idx += stride) {
    int e = (int)(idx & 15);
    int lane = (int)((idx >> 4) & 31);
    long long tile = idx >> 9;
    int tn = (int)(tile % ntiles);
    long long tk = tile / ntiles;
    long long k = tk * 32 + ((lane >> 4) << 4) + e;
    int n = tn * 16 + (lane & 15);
    dst[idx] = (k < K) ? (bf16)src[k * N + n] : (bf16)0.0f;
  }
}

__global__ void k_sdec(const float* __restrict__ S, const float* __restrict__ Wd,
                       float* __restrict__ sdec) {
  int b = blockIdx.x, col = threadIdx.x;
  float s = 0.f;
  for (int j = 0; j < RNN; ++j) s += S[b * RNN + j] * Wd[j * U_ + col];
  sdec[b * U_ + col] = s;
}

// ---------------- fused bah + attention-chain (WMMA) ----------------
__global__ void __launch_bounds__(128) k_bah_att(
    const float* __restrict__ h_enc, const bf16* __restrict__ Wenc_p,
    const float* __restrict__ sdec, const bf16* __restrict__ Wa1_p,
    const bf16* __restrict__ Wa2_p, const float* __restrict__ Wa3,
    float* __restrict__ bah, float* __restrict__ e_out) {
  __shared__ bf16  sA[16][DENC];   // h_enc tile (bf16)
  __shared__ bf16  sBah[16][U_];   // bah tile (bf16) for chain
  __shared__ bf16  sT1[16][A1_];
  __shared__ float sT2[16][A2_];

  const int tid  = threadIdx.x;
  const int lane = tid & 31;
  const int w    = tid >> 5;
  const size_t row0 = (size_t)blockIdx.x * 16;
  const int b = (int)(row0 / T_);
  const int mhi = (lane >> 4) << 3;  // C-frag row offset

  // stage h_enc tile as bf16 (float4 loads, b64 LDS stores)
  for (int idx = tid; idx < 16 * DENC / 4; idx += 128) {
    int r = idx >> 6, c4 = (idx & 63) << 2;
    const float4 v = *(const float4*)&h_enc[(row0 + r) * DENC + c4];
    PK4 pk;
    pk.h[0] = (bf16)v.x; pk.h[1] = (bf16)v.y;
    pk.h[2] = (bf16)v.z; pk.h[3] = (bf16)v.w;
    *(uint2*)&sA[r][c4] = pk.u;
  }
  __syncthreads();

  // bah tile: each wave owns 32 columns (2 WMMA n-tiles), K = 256 -> 8 steps
  v8f c0 = {}, c1 = {};
#pragma unroll
  for (int kk = 0; kk < 8; ++kk) {
    v16bf a  = load_a_frag(&sA[0][0], DENC, kk * 32, lane);
    v16bf b0 = load_b_packed(Wenc_p, 8, kk, w * 2, lane);
    v16bf b1 = load_b_packed(Wenc_p, 8, kk, w * 2 + 1, lane);
    c0 = wmma_bf16(a, b0, c0);
    c1 = wmma_bf16(a, b1, c1);
  }
  const int col0 = w * 32 + (lane & 15);
  const float sd0 = sdec[b * U_ + col0];
  const float sd1 = sdec[b * U_ + col0 + 16];
#pragma unroll
  for (int r = 0; r < 8; ++r) {
    int m = r + mhi;
    float v0 = c0[r] + sd0;
    float v1 = c1[r] + sd1;
    bah[(row0 + m) * U_ + col0]      = v0;
    bah[(row0 + m) * U_ + col0 + 16] = v1;
    sBah[m][col0]      = (bf16)v0;
    sBah[m][col0 + 16] = (bf16)v1;
  }
  __syncthreads();

  // t1 = tanh(bah @ W_att1) : 16x64, wave w -> n-tile w, K=128 -> 4 steps
  {
    v8f acc = {};
#pragma unroll
    for (int kk = 0; kk < 4; ++kk) {
      v16bf a  = load_a_frag(&sBah[0][0], U_, kk * 32, lane);
      v16bf bb = load_b_packed(Wa1_p, 4, kk, w, lane);
      acc = wmma_bf16(a, bb, acc);
    }
    int col = w * 16 + (lane & 15);
#pragma unroll
    for (int r = 0; r < 8; ++r) sT1[r + mhi][col] = (bf16)tanhf(acc[r]);
  }
  __syncthreads();

  // t2 = relu(t1 @ W_att2) : 16x32, waves 0..1, K=64 -> 2 steps
  if (w < 2) {
    v8f acc = {};
#pragma unroll
    for (int kk = 0; kk < 2; ++kk) {
      v16bf a  = load_a_frag(&sT1[0][0], A1_, kk * 32, lane);
      v16bf bb = load_b_packed(Wa2_p, 2, kk, w, lane);
      acc = wmma_bf16(a, bb, acc);
    }
    int col = w * 16 + (lane & 15);
#pragma unroll
    for (int r = 0; r < 8; ++r) sT2[r + mhi][col] = fmaxf(acc[r], 0.0f);
  }
  __syncthreads();

  // e = relu(t2 @ W_att3) : 16x1
  if (tid < 16) {
    float s = 0.f;
#pragma unroll
    for (int j = 0; j < A2_; ++j) s += sT2[tid][j] * Wa3[j];
    e_out[row0 + tid] = fmaxf(s, 0.0f);
  }
}

// ---------------- softmax over T per batch ----------------
__global__ void __launch_bounds__(256) k_softmax_T(const float* __restrict__ e,
                                                   float* __restrict__ aw) {
  __shared__ float red[256];
  int b = blockIdx.x, t = threadIdx.x;
  float mx = -3.4e38f;
  for (int i = t; i < T_; i += 256) mx = fmaxf(mx, e[b * T_ + i]);
  red[t] = mx;
  __syncthreads();
  for (int off = 128; off > 0; off >>= 1) {
    if (t < off) red[t] = fmaxf(red[t], red[t + off]);
    __syncthreads();
  }
  float m = red[0];
  __syncthreads();
  float s = 0.f;
  for (int i = t; i < T_; i += 256) s += expf(e[b * T_ + i] - m);
  red[t] = s;
  __syncthreads();
  for (int off = 128; off > 0; off >>= 1) {
    if (t < off) red[t] += red[t + off];
    __syncthreads();
  }
  float inv = 1.0f / red[0];
  for (int i = t; i < T_; i += 256) aw[b * T_ + i] = expf(e[b * T_ + i] - m) * inv;
}

// ---------------- conv1 (1->8 ch) + maxpool + tanh (tanh monotone => after pool) ----------------
__global__ void __launch_bounds__(128) k_conv1(const float* __restrict__ bah,
                                               const float* __restrict__ kw,
                                               const float* __restrict__ kb,
                                               float* __restrict__ c1) {
  __shared__ float sx[4][128];
  __shared__ float sw[72];
  __shared__ float sb[8];
  __shared__ float sconv[2][126][8];
  int tid = threadIdx.x;
  int n = blockIdx.x / 45;
  int i = blockIdx.x % 45;
  if (tid < 72) sw[tid] = kw[tid];
  if (tid < 8) sb[tid] = kb[tid];
  for (int idx = tid; idx < 512; idx += 128) {
    int r = idx >> 7, c = idx & 127;
    sx[r][c] = bah[((size_t)n * 48 + i + r) * 128 + c];
  }
  __syncthreads();
  for (int idx = tid; idx < 2 * 126 * 8; idx += 128) {
    int a = idx / (126 * 8);
    int rem = idx % (126 * 8);
    int j = rem >> 3, k = rem & 7;
    float v = sb[k];
#pragma unroll
    for (int di = 0; di < 3; ++di)
#pragma unroll
      for (int dj = 0; dj < 3; ++dj)
        v += sx[a + di][j + dj] * sw[(di * 3 + dj) * 8 + k];
    sconv[a][j][k] = v;
  }
  __syncthreads();
  for (int idx = tid; idx < 125 * 8; idx += 128) {
    int j = idx >> 3, k = idx & 7;
    float m = fmaxf(fmaxf(sconv[0][j][k], sconv[0][j + 1][k]),
                    fmaxf(sconv[1][j][k], sconv[1][j + 1][k]));
    c1[(((size_t)n * 45 + i) * 125 + j) * 8 + k] = tanhf(m);
  }
}

// ---------------- shared 8->8 ch conv + maxpool + tanh (conv2 / convy) ----------------
__global__ void __launch_bounds__(256) k_conv8(const float* __restrict__ in, int IH,
                                               int OI, const float* __restrict__ kw,
                                               const float* __restrict__ kb,
                                               float* __restrict__ out) {
  __shared__ float sx[4][125 * 8];
  __shared__ float sw[576];
  __shared__ float sb[8];
  __shared__ float sconv[2][123 * 8];
  int tid = threadIdx.x;
  int n = blockIdx.x / OI;
  int i = blockIdx.x % OI;
  if (tid < 8) sb[tid] = kb[tid];
  for (int idx = tid; idx < 576; idx += 256) sw[idx] = kw[idx];
  for (int idx = tid; idx < 4 * 1000; idx += 256) {
    int r = idx / 1000, c = idx % 1000;
    sx[r][c] = in[((size_t)n * IH + i + r) * 1000 + c];
  }
  __syncthreads();
  for (int idx = tid; idx < 2 * 984; idx += 256) {
    int a = idx / 984;
    int rem = idx % 984;
    int j = rem >> 3, k = rem & 7;
    float v = sb[k];
#pragma unroll
    for (int di = 0; di < 3; ++di)
#pragma unroll
      for (int dj = 0; dj < 3; ++dj) {
#pragma unroll
        for (int ci = 0; ci < 8; ++ci)
          v += sx[a + di][(j + dj) * 8 + ci] * sw[((di * 3 + dj) * 8 + ci) * 8 + k];
      }
    sconv[a][rem] = v;
  }
  __syncthreads();
  for (int idx = tid; idx < 122 * 8; idx += 256) {
    int j = idx >> 3, k = idx & 7;
    float m = fmaxf(fmaxf(sconv[0][j * 8 + k], sconv[0][(j + 1) * 8 + k]),
                    fmaxf(sconv[1][j * 8 + k], sconv[1][(j + 1) * 8 + k]));
    out[(((size_t)n * OI + i) * 122 + j) * 8 + k] = tanhf(m);
  }
}

// ---------------- monthly dense (WMMA, K=40992) + tanh + dense2 + relu ----------------
__global__ void __launch_bounds__(128) k_mdense(const float* __restrict__ c2,
                                                const bf16* __restrict__ Wm1p,
                                                const float* __restrict__ bm1,
                                                const float* __restrict__ Wm2,
                                                const float* __restrict__ bm2,
                                                float* __restrict__ em) {
  __shared__ bf16  sA[16][96];
  __shared__ float sT[16][64];
  const int tid = threadIdx.x, lane = tid & 31, w = tid >> 5;
  const int row0 = blockIdx.x * 16;
  const int mhi = (lane >> 4) << 3;
  v8f acc = {};
  for (int kc = 0; kc < FLATM; kc += 96) {  // 427 stages of 3 k-steps
    __syncthreads();
    for (int idx = tid; idx < 16 * 96 / 4; idx += 128) {
      int r = idx / 24, c4 = (idx % 24) << 2;
      const float4 v = *(const float4*)&c2[(size_t)(row0 + r) * FLATM + kc + c4];
      PK4 pk;
      pk.h[0] = (bf16)v.x; pk.h[1] = (bf16)v.y;
      pk.h[2] = (bf16)v.z; pk.h[3] = (bf16)v.w;
      *(uint2*)&sA[r][c4] = pk.u;
    }
    __syncthreads();
#pragma unroll
    for (int kk = 0; kk < 3; ++kk) {
      v16bf a  = load_a_frag(&sA[0][0], 96, kk * 32, lane);
      v16bf bb = load_b_packed(Wm1p, 4, (kc >> 5) + kk, w, lane);
      acc = wmma_bf16(a, bb, acc);
    }
  }
  int col = w * 16 + (lane & 15);
#pragma unroll
  for (int r = 0; r < 8; ++r) sT[r + mhi][col] = tanhf(acc[r] + bm1[col]);
  __syncthreads();
  if (tid < 16) {
    float s = bm2[0];
#pragma unroll
    for (int j = 0; j < 64; ++j) s += sT[tid][j] * Wm2[j];
    em[row0 + tid] = fmaxf(s, 0.0f);
  }
}

// ---------------- yearly dense partial (WMMA, K split deterministically) ----------------
__global__ void __launch_bounds__(128) k_ydense_partial(const float* __restrict__ cy,
                                                        const bf16* __restrict__ Wy1p,
                                                        float* __restrict__ eypart) {
  __shared__ bf16 sA[16][32];
  const int tid = threadIdx.x, lane = tid & 31, w = tid >> 5;
  const int row0 = blockIdx.x * 16;
  const int KSTEPS = FLATY_PAD / 32;  // 16379
  const int PER = (KSTEPS + YSPLIT - 1) / YSPLIT;
  int ks0 = blockIdx.y * PER;
  int ks1 = ks0 + PER;
  if (ks1 > KSTEPS) ks1 = KSTEPS;
  const int mhi = (lane >> 4) << 3;
  v8f acc = {};
  for (int ks = ks0; ks < ks1; ++ks) {
    int kc = ks * 32;
    __syncthreads();
    for (int idx = tid; idx < 128; idx += 128) {
      int r = idx >> 3, c4 = (idx & 7) << 2;
      int k = kc + c4;
      PK4 pk;
      if (k < FLATY) {
        const float4 v = *(const float4*)&cy[(size_t)(row0 + r) * FLATY + k];
        pk.h[0] = (bf16)v.x; pk.h[1] = (bf16)v.y;
        pk.h[2] = (bf16)v.z; pk.h[3] = (bf16)v.w;
      } else {
        pk.h[0] = pk.h[1] = pk.h[2] = pk.h[3] = (bf16)0.0f;
      }
      *(uint2*)&sA[r][c4] = pk.u;
    }
    __syncthreads();
    v16bf a  = load_a_frag(&sA[0][0], 32, 0, lane);
    v16bf bb = load_b_packed(Wy1p, 4, ks, w, lane);  // padded rows are zero
    acc = wmma_bf16(a, bb, acc);
  }
  int col = w * 16 + (lane & 15);
#pragma unroll
  for (int r = 0; r < 8; ++r)
    eypart[((size_t)blockIdx.y * 160 + row0 + r + mhi) * 64 + col] = acc[r];
}

__global__ void k_ey(const float* __restrict__ eypart, const float* __restrict__ by1,
                     const float* __restrict__ Wy2, const float* __restrict__ by2,
                     float* __restrict__ ey) {
  __shared__ float red[64];
  int s = blockIdx.x, j = threadIdx.x;
  float acc = 0.f;
  for (int p = 0; p < YSPLIT; ++p) acc += eypart[((size_t)p * 160 + s) * 64 + j];
  red[j] = tanhf(acc + by1[j]) * Wy2[j];
  __syncthreads();
  for (int off = 32; off > 0; off >>= 1) {
    if (j < off) red[j] += red[j + off];
    __syncthreads();
  }
  if (j == 0) ey[s] = fmaxf(red[0] + by2[0], 0.0f);
}

// ---------------- small-group softmax (12 months / 10 years) ----------------
__global__ void k_softmax_small(const float* __restrict__ in, float* __restrict__ out,
                                int glen) {
  __shared__ float sv[32];
  __shared__ float ssum;
  int g = blockIdx.x, t = threadIdx.x;
  if (t < glen) sv[t] = in[g * glen + t];
  __syncthreads();
  if (t == 0) {
    float mx = -3.4e38f;
    for (int i = 0; i < glen; ++i) mx = fmaxf(mx, sv[i]);
    float s = 0.f;
    for (int i = 0; i < glen; ++i) {
      sv[i] = expf(sv[i] - mx);
      s += sv[i];
    }
    ssum = s;
  }
  __syncthreads();
  if (t < glen) out[g * glen + t] = sv[t] / ssum;
}

// ---------------- hierarchical weight combine ----------------
__global__ void k_combine(const float* __restrict__ aw, const float* __restrict__ m_att,
                          const float* __restrict__ y_att, float* __restrict__ new_w) {
  int idx = blockIdx.x * blockDim.x + threadIdx.x;
  if (idx >= B_ * T_) return;
  int b = idx / T_, t = idx % T_;
  int y = t / 576, m = (t % 576) / 48;
  new_w[idx] = aw[idx] * m_att[(b * 10 + y) * 12 + m] * y_att[b * 10 + y];
}

// ---------------- context: ctx[b,d] = sum_t w[b,t] * h_enc[b,t,d] ----------------
__global__ void __launch_bounds__(256) k_ctx(const float* __restrict__ new_w,
                                             const float* __restrict__ h_enc,
                                             float* __restrict__ ctx) {
  int b = blockIdx.x, d = threadIdx.x;
  float acc = 0.f;
  for (int t = 0; t < T_; ++t)
    acc += new_w[b * T_ + t] * h_enc[((size_t)b * T_ + t) * DENC + d];
  ctx[b * DENC + d] = acc;
}

// ---------------- LSTM step ----------------
__global__ void __launch_bounds__(128) k_lstm_z(const float* __restrict__ ctx,
                                                const float* __restrict__ dec,
                                                const float* __restrict__ S,
                                                const float* __restrict__ Wl,
                                                const float* __restrict__ Ul,
                                                const float* __restrict__ bl,
                                                float* __restrict__ z) {
  int b = blockIdx.x;
  int col = blockIdx.y * 128 + threadIdx.x;
  float s = bl[col];
  for (int j = 0; j < DENC; ++j) s += ctx[b * DENC + j] * Wl[j * 1024 + col];
  for (int j = 0; j < FDEC; ++j) s += dec[b * FDEC + j] * Wl[(DENC + j) * 1024 + col];
  for (int j = 0; j < RNN; ++j) s += S[b * RNN + j] * Ul[j * 1024 + col];
  z[b * 1024 + col] = s;
}

__device__ inline float sigf(float x) { return 1.0f / (1.0f + expf(-x)); }

__global__ void k_lstm_gates(const float* __restrict__ z, const float* __restrict__ Cp,
                             float* __restrict__ out_h, float* __restrict__ out_c) {
  int b = blockIdx.x, j = threadIdx.x;
  float zi = z[b * 1024 + j];
  float zf = z[b * 1024 + 256 + j];
  float zg = z[b * 1024 + 512 + j];
  float zo = z[b * 1024 + 768 + j];
  float c = sigf(zf) * Cp[b * RNN + j] + sigf(zi) * tanhf(zg);
  float h = sigf(zo) * tanhf(c);
  out_h[b * RNN + j] = h;
  out_c[b * RNN + j] = c;
}

__global__ void k_outdense(const float* __restrict__ h, const float* __restrict__ Wo,
                           const float* __restrict__ bo, float* __restrict__ out) {
  int b = blockIdx.x, k = threadIdx.x;
  float s = bo[k];
  for (int j = 0; j < RNN; ++j) s += h[b * RNN + j] * Wo[j * NOUT + k];
  out[b * NOUT + k] = fmaxf(s, 0.0f);
}

// ---------------- host ----------------
extern "C" void kernel_launch(void* const* d_in, const int* in_sizes, int n_in,
                              void* d_out, int out_size, void* d_ws, size_t ws_size,
                              hipStream_t stream) {
  (void)in_sizes; (void)n_in; (void)out_size; (void)ws_size;
  const float* dec_input = (const float*)d_in[0];
  const float* S_prev    = (const float*)d_in[1];
  const float* C_prev    = (const float*)d_in[2];
  const float* h_enc     = (const float*)d_in[3];
  const float* W_enc     = (const float*)d_in[4];
  const float* W_dec     = (const float*)d_in[5];
  const float* W_att1    = (const float*)d_in[6];
  const float* W_att2    = (const float*)d_in[7];
  const float* W_att3    = (const float*)d_in[8];
  const float* conv1_k   = (const float*)d_in[9];
  const float* conv1_b   = (const float*)d_in[10];
  const float* conv2_k   = (const float*)d_in[11];
  const float* conv2_b   = (const float*)d_in[12];
  const float* convy_k   = (const float*)d_in[13];
  const float* convy_b   = (const float*)d_in[14];
  const float* Wm1       = (const float*)d_in[15];
  const float* bm1       = (const float*)d_in[16];
  const float* Wm2       = (const float*)d_in[17];
  const float* bm2       = (const float*)d_in[18];
  const float* Wy1       = (const float*)d_in[19];
  const float* by1       = (const float*)d_in[20];
  const float* Wy2       = (const float*)d_in[21];
  const float* by2       = (const float*)d_in[22];
  const float* W_lstm    = (const float*)d_in[23];
  const float* U_lstm    = (const float*)d_in[24];
  const float* b_lstm    = (const float*)d_in[25];
  const float* W_out     = (const float*)d_in[26];
  const float* b_out     = (const float*)d_in[27];

  float* out   = (float*)d_out;                 // [16,1,64]
  float* out_h = out + 1024;                    // [16,256]
  float* out_c = out + 5120;                    // [16,256]
  float* new_w = out + 9216;                    // [16,5760]

  char* base = (char*)d_ws;
  size_t off = 0;
  auto alloc = [&](size_t bytes) -> char* {
    off = (off + 255) & ~(size_t)255;
    char* p = base + off;
    off += bytes;
    return p;
  };
  float* bah    = (float*)alloc((size_t)92160 * 128 * 4);
  float* c1     = (float*)alloc((size_t)86400 * 1000 * 4);
  float* c2     = (float*)alloc((size_t)1920 * FLATM * 4);
  float* cy     = (float*)alloc((size_t)160 * FLATY * 4);
  bf16*  Wy1p   = (bf16*)alloc((size_t)FLATY_PAD * 64 * 2);
  bf16*  Wm1p   = (bf16*)alloc((size_t)FLATM * 64 * 2);
  bf16*  Wencp  = (bf16*)alloc((size_t)32768 * 2);
  bf16*  Wa1p   = (bf16*)alloc((size_t)8192 * 2);
  bf16*  Wa2p   = (bf16*)alloc((size_t)2048 * 2);
  float* sdec   = (float*)alloc(2048 * 4);
  float* e_buf  = (float*)alloc((size_t)92160 * 4);
  float* aw     = (float*)alloc((size_t)92160 * 4);
  float* em     = (float*)alloc(1920 * 4);
  float* m_att  = (float*)alloc(1920 * 4);
  float* eypart = (float*)alloc((size_t)YSPLIT * 160 * 64 * 4);
  float* ey     = (float*)alloc(160 * 4);
  float* y_att  = (float*)alloc(160 * 4);
  float* ctx    = (float*)alloc(4096 * 4);
  float* zbuf   = (float*)alloc(16384 * 4);

  // weight packing into WMMA-B fragment order (bf16)
  k_pack_b<<<128, 256, 0, stream>>>(W_enc, Wencp, 256, 128, 256);
  k_pack_b<<<32, 256, 0, stream>>>(W_att1, Wa1p, 128, 64, 128);
  k_pack_b<<<8, 256, 0, stream>>>(W_att2, Wa2p, 64, 32, 64);
  k_pack_b<<<4096, 256, 0, stream>>>(Wm1, Wm1p, FLATM, 64, FLATM);
  k_pack_b<<<8192, 256, 0, stream>>>(Wy1, Wy1p, FLATY, 64, FLATY_PAD);

  k_sdec<<<16, 128, 0, stream>>>(S_prev, W_dec, sdec);
  k_bah_att<<<T_ * B_ / 16, 128, 0, stream>>>(h_enc, Wencp, sdec, Wa1p, Wa2p, W_att3,
                                              bah, e_buf);
  k_softmax_T<<<16, 256, 0, stream>>>(e_buf, aw);

  k_conv1<<<1920 * 45, 128, 0, stream>>>(bah, conv1_k, conv1_b, c1);
  k_conv8<<<1920 * 42, 256, 0, stream>>>(c1, 45, 42, conv2_k, conv2_b, c2);
  k_mdense<<<120, 128, 0, stream>>>(c2, Wm1p, bm1, Wm2, bm2, em);
  k_softmax_small<<<160, 32, 0, stream>>>(em, m_att, 12);

  k_conv8<<<160 * 537, 256, 0, stream>>>(c1, 540, 537, convy_k, convy_b, cy);
  k_ydense_partial<<<dim3(10, YSPLIT), 128, 0, stream>>>(cy, Wy1p, eypart);
  k_ey<<<160, 64, 0, stream>>>(eypart, by1, Wy2, by2, ey);
  k_softmax_small<<<16, 32, 0, stream>>>(ey, y_att, 10);

  k_combine<<<(B_ * T_ + 255) / 256, 256, 0, stream>>>(aw, m_att, y_att, new_w);
  k_ctx<<<16, 256, 0, stream>>>(new_w, h_enc, ctx);
  k_lstm_z<<<dim3(16, 8), 128, 0, stream>>>(ctx, dec_input, S_prev, W_lstm, U_lstm,
                                            b_lstm, zbuf);
  k_lstm_gates<<<16, 256, 0, stream>>>(zbuf, C_prev, out_h, out_c);
  k_outdense<<<16, 64, 0, stream>>>(out_h, W_out, b_out, out);
}
